// CrossHeadAttention_49718541418686
// MI455X (gfx1250) — compile-verified, matrix operand order checked by reading, not run on hardware
//
#include <hip/hip_runtime.h>
#include <hip/hip_bf16.h>
#include <math.h>

typedef __bf16 bf16_t;
typedef __attribute__((ext_vector_type(16))) __bf16 v16bf;
typedef __attribute__((ext_vector_type(8)))  float  v8f;
typedef __attribute__((ext_vector_type(4)))  unsigned int u32x4;
typedef __attribute__((ext_vector_type(8)))  int i32x8;
typedef __attribute__((ext_vector_type(4)))  int i32x4;

#define S_LEN  2048
#define DMODEL 1024
#define NHEAD  16
#define HDIM   64
#define MTOT   4096   // B * S

#if defined(__has_builtin)
#if __has_builtin(__builtin_amdgcn_tensor_load_to_lds)
#define HAVE_TDM 1
#endif
#endif
#ifndef HAVE_TDM
#define HAVE_TDM 0
#endif

// ---------------------------------------------------------------------------
// Fragment helpers (CDNA5 WMMA 16-bit layouts, wave32)
//  A (16xK row-major) / Bt (N x K, one row per output column):
//    lane L: row/col = L&15, kbase = (L>>4)*8
//    v16bf elems j=0..7  -> K = kbase + j        (VGPRs 0..3)
//               j=8..15 -> K = 16 + kbase + j-8 (VGPRs 4..7)
//  C/D f32: elem r -> row M = (L>>4)*8 + r, col N = L&15
// ---------------------------------------------------------------------------
__device__ __forceinline__ v16bf load_frag16(const bf16_t* rowp, int kbase) {
  v16bf a;
  ((uint4*)&a)[0] = *(const uint4*)(rowp + kbase);        // K = kbase..kbase+7
  ((uint4*)&a)[1] = *(const uint4*)(rowp + 16 + kbase);   // K = 16+kbase..+7
  return a;
}

__device__ __forceinline__ v8f wmma_bf16(v16bf a, v16bf b, v8f c) {
  return __builtin_amdgcn_wmma_f32_16x16x32_bf16(
      false, a, false, b, (short)0, c, false, false);
}

__device__ __forceinline__ v8f zero8() {
  return (v8f){0.f,0.f,0.f,0.f,0.f,0.f,0.f,0.f};
}

// ---------------------------------------------------------------------------
// 32x64 wave GEMM core: acc[2][4] covers rows m0..m0+31, cols n0..n0+63.
// 12 b128 loads per 8 WMMAs (A fragments reused 4x, B fragments reused 2x).
// ---------------------------------------------------------------------------
__device__ __forceinline__ void wave_gemm_32x64(const bf16_t* __restrict__ X,
                                                const bf16_t* __restrict__ Wt,
                                                int m0, int n0, v8f acc[2][4]) {
  const int lane  = threadIdx.x & 31;
  const int r16   = lane & 15;
  const int kbase = (lane >> 4) << 3;
  const bf16_t* arow0 = X  + (size_t)(m0 + r16) * DMODEL;
  const bf16_t* arow1 = X  + (size_t)(m0 + 16 + r16) * DMODEL;
  const bf16_t* brow  = Wt + (size_t)(n0 + r16) * DMODEL;
  for (int k0 = 0; k0 < DMODEL; k0 += 32) {
    v16bf a0 = load_frag16(arow0 + k0, kbase);
    v16bf a1 = load_frag16(arow1 + k0, kbase);
#pragma unroll
    for (int c = 0; c < 4; ++c) {
      v16bf b = load_frag16(brow + (size_t)(c * 16) * DMODEL + k0, kbase);
      acc[0][c] = wmma_bf16(a0, b, acc[0][c]);
      acc[1][c] = wmma_bf16(a1, b, acc[1][c]);
    }
  }
}

// ---------------------------------------------------------------------------
// Conversions
// ---------------------------------------------------------------------------
__global__ void f32_to_bf16_kernel(const float* __restrict__ src,
                                   bf16_t* __restrict__ dst, int n) {
  int i = blockIdx.x * blockDim.x + threadIdx.x;
  if (i < n) dst[i] = (bf16_t)src[i];
}

// W [H][D][HD] f32 -> Wt [c = h*HD+hd][k = d] bf16 (k-contiguous rows)
__global__ void proj_w_transpose_kernel(const float* __restrict__ W,
                                        bf16_t* __restrict__ Wt) {
  int idx = blockIdx.x * blockDim.x + threadIdx.x;   // over 1024*1024
  int c = idx >> 10, k = idx & 1023;
  Wt[idx] = (bf16_t)W[((size_t)(c >> 6) * DMODEL + k) * HDIM + (c & 63)];
}

// Wo [D][D] f32 -> Wot[c][k] = Wo[k][c] bf16
__global__ void wo_transpose_kernel(const float* __restrict__ W,
                                    bf16_t* __restrict__ Wt) {
  int idx = blockIdx.x * blockDim.x + threadIdx.x;
  int c = idx >> 10, k = idx & 1023;
  Wt[idx] = (bf16_t)W[(size_t)k * DMODEL + c];
}

// ---------------------------------------------------------------------------
// QKV projection: 32x64 tile per wave; out[b,h,s,hd] bf16 = X @ Wt^T + bias
// ---------------------------------------------------------------------------
__global__ void __launch_bounds__(256)
qkv_proj_kernel(const bf16_t* __restrict__ X, const bf16_t* __restrict__ Wt,
                const float* __restrict__ bias, bf16_t* __restrict__ out) {
  const int wave = threadIdx.x >> 5;
  const int lane = threadIdx.x & 31;
  const int tile = blockIdx.x * 8 + wave;
  const int m0 = (tile >> 4) << 5;   // 128 m-tiles of 32
  const int n0 = (tile & 15) << 6;   // 16 n-tiles of 64

  v8f acc[2][4];
#pragma unroll
  for (int i = 0; i < 2; ++i)
#pragma unroll
    for (int c = 0; c < 4; ++c) acc[i][c] = zero8();

  wave_gemm_32x64(X, Wt, m0, n0, acc);

  const int r16 = lane & 15;
#pragma unroll
  for (int half = 0; half < 2; ++half) {
    const int mbase = m0 + half * 16 + ((lane >> 4) << 3);
#pragma unroll
    for (int c = 0; c < 4; ++c) {
      const int col = n0 + c * 16 + r16;
      const float bv = bias[col];
      const int h = col >> 6, hd = col & 63;
#pragma unroll
      for (int r = 0; r < 8; ++r) {
        const int m = mbase + r;
        const int b = m >> 11, s = m & 2047;
        out[(((size_t)(b * NHEAD + h) * S_LEN) + s) * HDIM + hd] =
            (bf16_t)(acc[half][c][r] + bv);
      }
    }
  }
}

// ---------------------------------------------------------------------------
// Output projection: out f32 [4096][1024] = A @ Wot^T + bo
// ---------------------------------------------------------------------------
__global__ void __launch_bounds__(256)
out_proj_kernel(const bf16_t* __restrict__ A, const bf16_t* __restrict__ Wot,
                const float* __restrict__ bo, float* __restrict__ out) {
  const int wave = threadIdx.x >> 5;
  const int lane = threadIdx.x & 31;
  const int tile = blockIdx.x * 8 + wave;
  const int m0 = (tile >> 4) << 5;
  const int n0 = (tile & 15) << 6;

  v8f acc[2][4];
#pragma unroll
  for (int i = 0; i < 2; ++i)
#pragma unroll
    for (int c = 0; c < 4; ++c) acc[i][c] = zero8();

  wave_gemm_32x64(A, Wot, m0, n0, acc);

  const int r16 = lane & 15;
#pragma unroll
  for (int half = 0; half < 2; ++half) {
    const int mbase = m0 + half * 16 + ((lane >> 4) << 3);
#pragma unroll
    for (int c = 0; c < 4; ++c) {
      const int col = n0 + c * 16 + r16;
      const float bv = bo[col];
#pragma unroll
      for (int r = 0; r < 8; ++r)
        out[(size_t)(mbase + r) * DMODEL + col] = acc[half][c][r] + bv;
    }
  }
}

// ---------------------------------------------------------------------------
// TDM: 2D tile (rows x cols bf16) -> LDS. D# per CDNA5 ISA ch.8.
// Wave-uniform args; EXEC ignored by TDM; completion via TENSORcnt (in-order).
// ---------------------------------------------------------------------------
#if HAVE_TDM
__device__ __forceinline__ void tdm_load_2d(const bf16_t* gsrc, unsigned lds_off,
                                            int rows, int cols, int gstride) {
  unsigned long long ga = (unsigned long long)(uintptr_t)gsrc;
  u32x4 g0 = { 1u,                                   // count=1, no gather
               lds_off,                              // LDS byte address
               (unsigned)(ga & 0xFFFFFFFFu),         // global_addr[31:0]
               ((unsigned)((ga >> 32) & 0x1FFFFFFu)) // global_addr[56:32]
                 | 0x80000000u };                    // type=2 ("image")
  i32x8 g1 = { 0x00010000,                           // data_size=1 (2 bytes)
               cols << 16,                           // tensor_dim0[15:0]
               rows << 16,                           // tensor_dim1[15:0]
               cols << 16,                           // tile_dim0
               rows,                                 // tile_dim1 (tile_dim2=0)
               gstride,                              // tensor_dim0_stride[31:0]
               0, 0 };
  i32x4 gz = { 0, 0, 0, 0 };
#if __clang_major__ >= 23
  i32x8 gz8 = { 0, 0, 0, 0, 0, 0, 0, 0 };
  __builtin_amdgcn_tensor_load_to_lds(g0, g1, gz, gz, gz8, 0);
#else
  __builtin_amdgcn_tensor_load_to_lds(g0, g1, gz, gz, 0);
#endif
}
#endif

// ---------------------------------------------------------------------------
// Flash attention: block = 8 waves = 128 query rows of one (b,h).
// TDM double-buffers K/V tiles [32][64] into LDS (wave 0 issues next tile,
// s_wait_tensorcnt(2) guarantees current tile landed, in-order TENSORcnt).
// Block-cooperative LDS transpose of V -> b128 B-fragment loads for PV.
// ---------------------------------------------------------------------------
__global__ void __launch_bounds__(256)
attn_kernel(const bf16_t* __restrict__ Q, const bf16_t* __restrict__ Kmat,
            const bf16_t* __restrict__ V, const int* __restrict__ mask,
            bf16_t* __restrict__ attn_out) {
#if HAVE_TDM
  __shared__ __align__(16) bf16_t Kraw[2][32 * 64];  // [s_local][d]
  __shared__ __align__(16) bf16_t Vraw[2][32 * 64];  // [s_local][d]
  __shared__ __align__(16) bf16_t VT[2][64 * 32];    // [d][s_local]
#else
  __shared__ __align__(16) bf16_t Kraw[1][32 * 64];
  __shared__ __align__(16) bf16_t VT[1][64 * 32];
#endif
  __shared__ __align__(16) bf16_t Pst[8][16 * 32];   // per-wave P staging

  const int b    = blockIdx.z;
  const int h    = blockIdx.y;
  const int wave = threadIdx.x >> 5;
  const int lane = threadIdx.x & 31;
  const int q0   = blockIdx.x * 128 + wave * 16;

  const size_t bh = (size_t)(b * NHEAD + h) * S_LEN * HDIM;
  const bf16_t* Qbh = Q    + bh;
  const bf16_t* Kbh = Kmat + bh;
  const bf16_t* Vbh = V    + bh;

  const int r16   = lane & 15;
  const int kbase = (lane >> 4) << 3;
  const int mbase = (lane >> 4) << 3;

  // Q fragment (16 rows x 64 k), reused for the whole key loop
  const bf16_t* qrow = Qbh + (size_t)(q0 + r16) * HDIM;
  const v16bf qa0 = load_frag16(qrow, kbase);        // k = 0..31
  const v16bf qa1 = load_frag16(qrow + 32, kbase);   // k = 32..63

  float mrow[8], lrow[8];
  v8f oacc[4];
#pragma unroll
  for (int r = 0; r < 8; ++r) { mrow[r] = -INFINITY; lrow[r] = 0.f; }
#pragma unroll
  for (int c2 = 0; c2 < 4; ++c2) oacc[c2] = zero8();

#if HAVE_TDM
  // prologue: stage tile 0 into buffer 0
  if (wave == 0) {
    tdm_load_2d(Kbh, (unsigned)(uintptr_t)&Kraw[0][0], 32, 64, 64);
    tdm_load_2d(Vbh, (unsigned)(uintptr_t)&Vraw[0][0], 32, 64, 64);
  }
#endif

  int step = 0;
  for (int s0 = 0; s0 < S_LEN; s0 += 32, ++step) {
#if HAVE_TDM
    const int cur = step & 1;
    const int nxt = cur ^ 1;
    if (wave == 0) {
      if (s0 + 32 < S_LEN) {
        // issue next tile into the buffer last read in step-1 (safe: we are
        // past that step's trailing barrier)
        tdm_load_2d(Kbh + (size_t)(s0 + 32) * HDIM,
                    (unsigned)(uintptr_t)&Kraw[nxt][0], 32, 64, 64);
        tdm_load_2d(Vbh + (size_t)(s0 + 32) * HDIM,
                    (unsigned)(uintptr_t)&Vraw[nxt][0], 32, 64, 64);
        // 2 outstanding allowed (next tile); current tile's 2 ops are done
        __builtin_amdgcn_s_wait_tensorcnt(2);
      } else {
        __builtin_amdgcn_s_wait_tensorcnt(0);
      }
    }
    __syncthreads();   // current raw K/V tiles visible to all waves

    // cooperative transpose V: 1 b128 read + 8 b16 stores per thread
    {
      const int idx = threadIdx.x * 8;          // 0..2047
      const int sl = idx >> 6, d0 = idx & 63;
      uint4 raw = *(const uint4*)&Vraw[cur][idx];
      const bf16_t* e = (const bf16_t*)&raw;
#pragma unroll
      for (int j = 0; j < 8; ++j) VT[cur][(d0 + j) * 32 + sl] = e[j];
    }
    __syncthreads();   // VT ready
#else
    const int cur = 0;
    {
      const int idx = threadIdx.x * 8;
      const int sl = idx >> 6, d0 = idx & 63;
      *(uint4*)&Kraw[0][idx] = *(const uint4*)(Kbh + (size_t)(s0 + sl) * HDIM + d0);
      uint4 raw = *(const uint4*)(Vbh + (size_t)(s0 + sl) * HDIM + d0);
      const bf16_t* e = (const bf16_t*)&raw;
#pragma unroll
      for (int j = 0; j < 8; ++j) VT[0][(d0 + j) * 32 + sl] = e[j];
    }
    __syncthreads();
#endif
    const bf16_t* Kc  = &Kraw[cur][0];
    const bf16_t* VTc = &VT[cur][0];

    // ---- scores S = Q (16x64) @ K_tile^T (64x32), scale + mask ----
    float sc[2][8];
#pragma unroll
    for (int kb = 0; kb < 2; ++kb) {
      const bf16_t* krow = Kc + (kb * 16 + r16) * 64;   // B: col = key pos
      v16bf b0 = load_frag16(krow, kbase);
      v16bf b1 = load_frag16(krow + 32, kbase);
      v8f sacc = zero8();
      sacc = wmma_bf16(qa0, b0, sacc);
      sacc = wmma_bf16(qa1, b1, sacc);
      const int scol = s0 + kb * 16 + r16;
#pragma unroll
      for (int r = 0; r < 8; ++r) {
        const float vsc = sacc[r] * 0.125f;   // 1/sqrt(64)
        const int mk = mask[((size_t)b * S_LEN + (q0 + mbase + r)) * S_LEN + scol];
        sc[kb][r] = (mk == 0) ? -INFINITY : vsc;
      }
    }

    // ---- online softmax (rows live inside one 16-lane half-wave) ----
#pragma unroll
    for (int r = 0; r < 8; ++r) {
      float rmax = fmaxf(sc[0][r], sc[1][r]);
#pragma unroll
      for (int off = 8; off > 0; off >>= 1)
        rmax = fmaxf(rmax, __shfl_xor(rmax, off, 16));
      const float mnew  = fmaxf(mrow[r], rmax);
      const float alpha = __expf(mrow[r] - mnew);
      const float p0 = __expf(sc[0][r] - mnew);
      const float p1 = __expf(sc[1][r] - mnew);
      float rsum = p0 + p1;
#pragma unroll
      for (int off = 8; off > 0; off >>= 1)
        rsum += __shfl_xor(rsum, off, 16);
      lrow[r] = lrow[r] * alpha + rsum;
      mrow[r] = mnew;
#pragma unroll
      for (int c2 = 0; c2 < 4; ++c2) oacc[c2][r] *= alpha;
      Pst[wave][(mbase + r) * 32 + r16]      = (bf16_t)p0;
      Pst[wave][(mbase + r) * 32 + 16 + r16] = (bf16_t)p1;
    }

    // ---- O += P (16x32) @ V_tile (32x64) ----
    {
      const bf16_t* prow = &Pst[wave][r16 * 32];
      v16bf pa = load_frag16(prow, kbase);     // K = 0..31 (32 keys)
#pragma unroll
      for (int c2 = 0; c2 < 4; ++c2) {
        // B[k=s][n=d] = VT[d][s]; rows of VT are contiguous -> b128 reads
        const bf16_t* vrow = VTc + (c2 * 16 + r16) * 32;
        v16bf vb = load_frag16(vrow, kbase);
        oacc[c2] = wmma_bf16(pa, vb, oacc[c2]);
      }
    }
    __syncthreads();   // all waves done with current buffers
  }

  // ---- epilogue: normalize, concat heads along D ----
#pragma unroll
  for (int r = 0; r < 8; ++r) {
    const float inv = 1.0f / lrow[r];
    const int qg = q0 + mbase + r;
#pragma unroll
    for (int c2 = 0; c2 < 4; ++c2) {
      const int dcol = h * HDIM + c2 * 16 + r16;
      attn_out[((size_t)b * S_LEN + qg) * DMODEL + dcol] = (bf16_t)(oacc[c2][r] * inv);
    }
  }
}

// ---------------------------------------------------------------------------
// Host-side launch
// ---------------------------------------------------------------------------
extern "C" void kernel_launch(void* const* d_in, const int* in_sizes, int n_in,
                              void* d_out, int out_size, void* d_ws, size_t ws_size,
                              hipStream_t stream) {
  (void)in_sizes; (void)n_in; (void)out_size; (void)ws_size;
  const float* dec  = (const float*)d_in[0];
  const float* enc  = (const float*)d_in[1];
  const int*   mask = (const int*)  d_in[2];
  const float* Wq   = (const float*)d_in[3];
  const float* bq   = (const float*)d_in[4];
  const float* Wk   = (const float*)d_in[5];
  const float* bk   = (const float*)d_in[6];
  const float* Wv   = (const float*)d_in[7];
  const float* bv   = (const float*)d_in[8];
  const float* Wo   = (const float*)d_in[9];
  const float* bo   = (const float*)d_in[10];
  float* out = (float*)d_out;

  char* ws = (char*)d_ws;
  size_t off = 0;
  auto alloc = [&](size_t bytes) -> char* {
    char* p = ws + off;
    off += (bytes + 255) & ~(size_t)255;
    return p;
  };

  const size_t HS_ELEMS  = (size_t)MTOT * DMODEL;             // 4194304
  const size_t W_ELEMS   = (size_t)DMODEL * DMODEL;           // 1048576
  const size_t QKV_ELEMS = (size_t)2 * NHEAD * S_LEN * HDIM;  // 4194304

  bf16_t* enc_bf  = (bf16_t*)alloc(HS_ELEMS * 2);
  bf16_t* dec_bf  = (bf16_t*)alloc(HS_ELEMS * 2);
  bf16_t* Wqt     = (bf16_t*)alloc(W_ELEMS * 2);
  bf16_t* Wkt     = (bf16_t*)alloc(W_ELEMS * 2);
  bf16_t* Wvt     = (bf16_t*)alloc(W_ELEMS * 2);
  bf16_t* Wot     = (bf16_t*)alloc(W_ELEMS * 2);
  bf16_t* qbuf    = (bf16_t*)alloc(QKV_ELEMS * 2);
  bf16_t* kbuf    = (bf16_t*)alloc(QKV_ELEMS * 2);
  bf16_t* vbuf    = (bf16_t*)alloc(QKV_ELEMS * 2);
  bf16_t* attn_bf = (bf16_t*)alloc(HS_ELEMS * 2);

  // 1) fp32 -> bf16 conversions / weight transposes
  f32_to_bf16_kernel<<<(int)(HS_ELEMS / 256), 256, 0, stream>>>(enc, enc_bf, (int)HS_ELEMS);
  f32_to_bf16_kernel<<<(int)(HS_ELEMS / 256), 256, 0, stream>>>(dec, dec_bf, (int)HS_ELEMS);
  proj_w_transpose_kernel<<<(int)(W_ELEMS / 256), 256, 0, stream>>>(Wq, Wqt);
  proj_w_transpose_kernel<<<(int)(W_ELEMS / 256), 256, 0, stream>>>(Wk, Wkt);
  proj_w_transpose_kernel<<<(int)(W_ELEMS / 256), 256, 0, stream>>>(Wv, Wvt);
  wo_transpose_kernel<<<(int)(W_ELEMS / 256), 256, 0, stream>>>(Wo, Wot);

  // 2) projections (reference arg swap: q <- enc, k/v <- dec)
  qkv_proj_kernel<<<256, 256, 0, stream>>>(enc_bf, Wqt, bq, qbuf);
  qkv_proj_kernel<<<256, 256, 0, stream>>>(dec_bf, Wkt, bk, kbuf);
  qkv_proj_kernel<<<256, 256, 0, stream>>>(dec_bf, Wvt, bv, vbuf);

  // 3) attention: grid (q-blocks of 128, H, B)
  attn_kernel<<<dim3(S_LEN / 128, NHEAD, 2), 256, 0, stream>>>(qbuf, kbuf, vbuf, mask, attn_bf);

  // 4) output projection -> f32
  out_proj_kernel<<<256, 256, 0, stream>>>(attn_bf, Wot, bo, out);
}
